// PaiNN_9191230013851
// MI455X (gfx1250) — matrix operand order repaired and checked by reading, not compiled, not run on hardware
//
#include <hip/hip_runtime.h>
#include <math.h>

typedef __attribute__((ext_vector_type(16))) _Float16 v16h;
typedef __attribute__((ext_vector_type(8)))  float    v8f;

#define BMOL 512
#define MAT  24
#define FD   128
#define NRBF 20

// ---- shared memory layout (byte offsets), total 59008 B ----
// pool [0,36864): stages<=GEMM2: sF16[32][128] @0 (8192), hF16[32][128] @8192 (8192)
//                 stages>=pairs: rbf16[24][24][32] @0 (36864)  (K: 20 rbf*mask, [20]=mask, rest 0)
// phiF16[24][256] @36864 (12288) | rhat[24][24][3] f32 @49152 (6912)
// pos[72] @56064 | b1[128] @56352 | b2m[256] @56864 | bfm[256] @57888 | atomIdx[24] @58912
#define SMEM_BYTES 59008

#define WMMA_F16(a, b, c) \
    __builtin_amdgcn_wmma_f32_16x16x32_f16(false, (a), false, (b), (short)0, (c), false, false)

// A fragment (16x32, f16) from row-major LDS, stride in halfs.
__device__ __forceinline__ v16h ldsAfrag(const _Float16* p, int stride, int lane) {
    v16h a;
    int m = lane & 15;
    int koff = (lane & 16) ? 8 : 0;
#pragma unroll
    for (int e = 0; e < 16; ++e) {
        int k = ((e >> 3) << 4) | (koff + (e & 7));
        a[e] = p[m * stride + k];
    }
    return a;
}

// B fragment (32x16) from row-major f32 global, converting to f16.
__device__ __forceinline__ v16h gblBfragF32(const float* p, int stride, int lane) {
    v16h b;
    int n = lane & 15;
    int kb = (lane & 16) ? 8 : 0;
#pragma unroll
    for (int e = 0; e < 16; ++e) b[e] = (_Float16)p[(((e >> 3) << 4) + kb + (e & 7)) * stride + n];
    return b;
}

// A fragment for edge GEMM: rbf16[i][j][0..31] (stride 32), rows j padded 24->32 with zeros.
__device__ __forceinline__ v16h rbfAfrag(const _Float16* rbf, int i, int rt, int lane) {
    v16h a;
    int m = lane & 15;
    int j = rt * 16 + m;
    int koff = (lane & 16) ? 8 : 0;
    bool jok = (j < MAT);
    const _Float16* row = rbf + (i * MAT + (jok ? j : 0)) * 32;
#pragma unroll
    for (int e = 0; e < 16; ++e) {
        int k = ((e >> 3) << 4) | (koff + (e & 7));
        a[e] = jok ? row[k] : (_Float16)0.f;
    }
    return a;
}

__global__ __launch_bounds__(256) void painn_fused_kernel(
    const int* __restrict__ atoms, const float* __restrict__ apos,
    const float* __restrict__ emb, const float* __restrict__ W1,
    const float* __restrict__ b1, const float* __restrict__ W2,
    const float* __restrict__ b2, const float* __restrict__ Wf,
    const float* __restrict__ bf, float* __restrict__ out) {

    __shared__ __align__(16) unsigned char smem[SMEM_BYTES];
    _Float16* sF16  = (_Float16*)(smem);            // [32][128]
    _Float16* hF16  = (_Float16*)(smem + 8192);     // [32][128]
    _Float16* rbf16 = (_Float16*)(smem);            // [24][24][32] (reuses s/h region)
    _Float16* phiS  = (_Float16*)(smem + 36864);    // [24][256] (thirds 0 and 2)
    float* rhatS = (float*)(smem + 49152);          // [24][24][3]
    float* posS  = (float*)(smem + 56064);          // [24][3]
    float* b1S   = (float*)(smem + 56352);          // [128]
    float* b2S   = (float*)(smem + 56864);          // [256] mapped cols {0..127,256..383}
    float* bfS   = (float*)(smem + 57888);          // [256] same mapping
    int*   aIdx  = (int*)(smem + 58912);            // [24]

    const int b    = blockIdx.x;
    const int tid  = threadIdx.x;
    const int lane = tid & 31;
    const int wave = tid >> 5;

    // ---- stage 0: warm L2 for shared weights (global_prefetch_b8), zero-init,
    //      and load per-molecule scalars ----
    {   // W1: 64KB (1 line/thread), W2: 192KB (3/thread), Wf: 30KB (1 for tid<120)
        __builtin_prefetch((const char*)W1 + tid * 256, 0, 1);
#pragma unroll
        for (int q = 0; q < 3; ++q)
            __builtin_prefetch((const char*)W2 + (q * 256 + tid) * 256, 0, 1);
        if (tid < 120) __builtin_prefetch((const char*)Wf + tid * 256, 0, 1);
    }
    for (int idx = tid; idx < 4096; idx += 256) {
        sF16[idx] = (_Float16)0.f;
        hF16[idx] = (_Float16)0.f;
    }
    if (tid < MAT) aIdx[tid] = atoms[b * MAT + tid];
    if (tid < 72)  posS[tid] = apos[b * 72 + tid];
    if (tid < 128) b1S[tid] = b1[tid];
    {   // mapped biases (dead middle third skipped)
        int g = (tid < 128) ? tid : tid + 128;
        b2S[tid] = b2[g];
        bfS[tid] = bf[g];
    }
    __syncthreads();

    // ---- stage 1: gather node embeddings s = emb[atoms] ----
    for (int idx = tid; idx < MAT * FD; idx += 256) {
        int row = idx >> 7, col = idx & 127;
        sF16[row * FD + col] = (_Float16)emb[aIdx[row] * FD + col];
    }
    __syncthreads();

    // ---- stage 2: GEMM1  h = silu(s @ W1 + b1)  [24x128]x[128x128] ----
    for (int t = wave; t < 16; t += 8) {           // uniform: 2 tiles/wave
        int rt = t >> 3, nt = t & 7;
        v8f c = {};
#pragma unroll
        for (int kt = 0; kt < 4; ++kt) {
            v16h a  = ldsAfrag(sF16 + (rt * 16) * FD + kt * 32, FD, lane);
            v16h bv = gblBfragF32(W1 + (kt * 32) * FD + nt * 16, FD, lane);
            c = WMMA_F16(a, bv, c);
        }
        int n = lane & 15, off = (lane & 16) ? 8 : 0;
        int col = nt * 16 + n;
        float bb = b1S[col];
#pragma unroll
        for (int k = 0; k < 8; ++k) {
            int row = rt * 16 + off + k;
            float x = c[k] + bb;
            // silu via raw v_rcp_f32 (single TRANS op, no IEEE div chain)
            float sg = __builtin_amdgcn_rcpf(1.f + __expf(-x));
            hF16[row * FD + col] = (_Float16)(x * sg);
        }
    }
    __syncthreads();

    // ---- stage 3: GEMM2  phi = h @ W2 + b2 (live thirds only) ----
    for (int t = wave; t < 32; t += 8) {           // uniform: 4 tiles/wave
        int rt = t >> 4, ct = t & 15;
        int gcolBase = (ct < 8) ? ct * 16 : ct * 16 + 128;   // skip dead cols 128..255
        v8f c = {};
#pragma unroll
        for (int kt = 0; kt < 4; ++kt) {
            v16h a  = ldsAfrag(hF16 + (rt * 16) * FD + kt * 32, FD, lane);
            v16h bv = gblBfragF32(W2 + (kt * 32) * 384 + gcolBase, 384, lane);
            c = WMMA_F16(a, bv, c);
        }
        int n = lane & 15, off = (lane & 16) ? 8 : 0;
        int pc = ct * 16 + n;
        float bb = b2S[pc];
#pragma unroll
        for (int k = 0; k < 8; ++k) {
            int row = rt * 16 + off + k;
            if (row < MAT) phiS[row * 256 + pc] = (_Float16)(c[k] + bb);
        }
    }
    __syncthreads();

    // ---- stage 4: pair geometry + masked RBF (overwrites s/h region) ----
    // rbf16[p][k] = mask * sin((k+1)*pi/5*d)/d for k<20, [20] = mask, [21..31] = 0
    for (int p = tid; p < MAT * MAT; p += 256) {
        int i = p / MAT, j = p - i * MAT;
        float dx = posS[i * 3 + 0] - posS[j * 3 + 0];
        float dy = posS[i * 3 + 1] - posS[j * 3 + 1];
        float dz = posS[i * 3 + 2] - posS[j * 3 + 2];
        float d  = __builtin_amdgcn_sqrtf(dx * dx + dy * dy + dz * dz);
        bool  mk = (d < 5.0f) && (i != j);
        float mskf = mk ? 1.f : 0.f;
        float dsafe = mk ? d : 1.0f;
        float inv = __builtin_amdgcn_rcpf(dsafe);
        rhatS[p * 3 + 0] = dx * inv;
        rhatS[p * 3 + 1] = dy * inv;
        rhatS[p * 3 + 2] = dz * inv;
        float w = 0.62831853071795864769f * dsafe;   // pi/5 * d
        float minv = mskf * inv;
        _Float16* row = rbf16 + p * 32;
#pragma unroll
        for (int nn = 1; nn <= NRBF; ++nn)
            row[nn - 1] = (_Float16)(__sinf((float)nn * w) * minv);
        row[NRBF] = (_Float16)mskf;                  // folds mask*bf via Wf-row 20 = bf
#pragma unroll
        for (int k = NRBF + 1; k < 32; ++k) row[k] = (_Float16)0.f;
    }
    __syncthreads();

    // ---- stage 5: edge GEMM + fused phi modulation + j-reduction ----
    // Each wave owns feature block f in [wave*16, wave*16+16) for BOTH channels.
    {
        int n = lane & 15, off = (lane & 16) ? 8 : 0;
        int kb = (lane & 16) ? 8 : 0;
        int pc0 = wave * 16 + n;      // scalar-channel phi col (global col pc0)
        int pc1 = pc0 + 128;          // vector-channel phi col (global col pc1+128)
        // Hoisted B fragments: rows 0..19 = Wf, row 20 = bf, rest 0.
        v16h bv0, bv1;
#pragma unroll
        for (int e = 0; e < 16; ++e) {
            int k = ((e >> 3) << 4) + kb + (e & 7);
            bv0[e] = (k < NRBF)  ? (_Float16)Wf[k * 384 + pc0]
                   : (k == NRBF) ? (_Float16)bfS[pc0] : (_Float16)0.f;
            bv1[e] = (k < NRBF)  ? (_Float16)Wf[k * 384 + (pc1 + 128)]
                   : (k == NRBF) ? (_Float16)bfS[pc1] : (_Float16)0.f;
        }

        for (int i = 0; i < MAT; ++i) {            // wave-uniform loop
            v16h a0 = rbfAfrag(rbf16, i, 0, lane); // neighbor rows 0..15
            v16h a1 = rbfAfrag(rbf16, i, 1, lane); // neighbor rows 16..31 (24+ zero)
            v8f z = {};
            v8f c00 = WMMA_F16(a0, bv0, z);
            v8f c01 = WMMA_F16(a1, bv0, z);
            v8f c10 = WMMA_F16(a0, bv1, z);
            v8f c11 = WMMA_F16(a1, bv1, z);

            float p = 0.f, px = 0.f, py = 0.f, pz = 0.f;
#pragma unroll
            for (int k = 0; k < 8; ++k) {
                int j0 = off + k;                  // rows of c?0: 0..15
                const float* rh0 = rhatS + (i * MAT + j0) * 3;
                p += c00[k] * (float)phiS[j0 * 256 + pc0];
                float w0 = c10[k] * (float)phiS[j0 * 256 + pc1];
                px += w0 * rh0[0];
                py += w0 * rh0[1];
                pz += w0 * rh0[2];
                int j1 = j0 + 16;                  // rows of c?1: 16..31
                int jc = (j1 < MAT) ? j1 : 0;      // c?1[k]==0 for j1>=24 (zero A rows)
                const float* rh1 = rhatS + (i * MAT + jc) * 3;
                p += c01[k] * (float)phiS[jc * 256 + pc0];
                float w1 = c11[k] * (float)phiS[jc * 256 + pc1];
                px += w1 * rh1[0];
                py += w1 * rh1[1];
                pz += w1 * rh1[2];
            }
            p  += __shfl_xor(p, 16, 32);
            px += __shfl_xor(px, 16, 32);
            py += __shfl_xor(py, 16, 32);
            pz += __shfl_xor(pz, 16, 32);
            if ((lane & 16) == 0) {
                size_t obase = (size_t)(b * MAT + i) * 512;
                int col = wave * 16 + lane;        // = f index for both channels
                out[obase + col] = emb[aIdx[i] * FD + col] + p;
                size_t vb = obase + 128 + (size_t)col * 3;
                out[vb + 0] = px;
                out[vb + 1] = py;
                out[vb + 2] = pz;
            }
        }
    }
}

extern "C" void kernel_launch(void* const* d_in, const int* in_sizes, int n_in,
                              void* d_out, int out_size, void* d_ws, size_t ws_size,
                              hipStream_t stream) {
    (void)in_sizes; (void)n_in; (void)out_size; (void)d_ws; (void)ws_size;
    const int*   atoms = (const int*)d_in[0];
    const float* apos  = (const float*)d_in[1];
    // d_in[2] = graph_indexes (unused: fixed [B,M] layout)
    const float* emb = (const float*)d_in[3];
    const float* W1  = (const float*)d_in[4];
    const float* b1  = (const float*)d_in[5];
    const float* W2  = (const float*)d_in[6];
    const float* b2  = (const float*)d_in[7];
    const float* Wf  = (const float*)d_in[8];
    const float* bf  = (const float*)d_in[9];
    float* out = (float*)d_out;

    painn_fused_kernel<<<BMOL, 256, 0, stream>>>(atoms, apos, emb, W1, b1, W2, b2, Wf, bf, out);
}
